// MaskedSelfAttentionHead_2259152797991
// MI455X (gfx1250) — matrix operand order repaired and checked by reading, not compiled, not run on hardware
//
#include <hip/hip_runtime.h>

#define BB   4
#define TT   4096
#define CC   1024
#define HSZ  64
#define NROWS (BB * TT)   // 16384

typedef __bf16          bf16x16 __attribute__((ext_vector_type(16)));
typedef __bf16          bf16x2  __attribute__((ext_vector_type(2)));
typedef float           f32x8   __attribute__((ext_vector_type(8)));
typedef unsigned int    u32x2   __attribute__((ext_vector_type(2)));
typedef unsigned int    u32x4   __attribute__((ext_vector_type(4)));
typedef unsigned int    u32x8   __attribute__((ext_vector_type(8)));

// Pack two f32 -> packed bf16 pair (RNE).  v_cvt_pk_bf16_f32 when available.
#if __has_builtin(__builtin_amdgcn_cvt_pk_bf16_f32)
__device__ __forceinline__ unsigned pack2(float a, float b) {
  bf16x2 v = __builtin_amdgcn_cvt_pk_bf16_f32(a, b);
  return __builtin_bit_cast(unsigned, v);
}
#else
__device__ __forceinline__ unsigned short f2bf_1(float f) {
  unsigned u = __builtin_bit_cast(unsigned, f);
  return (unsigned short)((u + 0x7FFFu + ((u >> 16) & 1u)) >> 16);
}
__device__ __forceinline__ unsigned pack2(float a, float b) {
  return (unsigned)f2bf_1(a) | ((unsigned)f2bf_1(b) << 16);
}
#endif

// Store two bf16 values to two (possibly non-adjacent) u16 slots:
// low half + high half (compiler uses ds/global *_d16_hi for the second).
__device__ __forceinline__ void store2bf(unsigned short* a0, unsigned short* a1,
                                         float x, float y) {
  unsigned p = pack2(x, y);
  *a0 = (unsigned short)p;
  *a1 = (unsigned short)(p >> 16);
}

__device__ __forceinline__ f32x8 zero8() {
  f32x8 z = {0.f, 0.f, 0.f, 0.f, 0.f, 0.f, 0.f, 0.f};
  return z;
}

__device__ __forceinline__ f32x8 wmma_bf16(bf16x16 a, bf16x16 b, f32x8 c) {
  return __builtin_amdgcn_wmma_f32_16x16x32_bf16(false, a, false, b, (short)0, c,
                                                 false, false);
}

// Build a 16-element bf16 fragment from two 16-byte contiguous runs.
__device__ __forceinline__ bf16x16 frag_ld(const unsigned short* p0,
                                           const unsigned short* p1) {
  u32x4 a = *(const u32x4*)p0;
  u32x4 b = *(const u32x4*)p1;
  u32x8 t;
  t[0] = a[0]; t[1] = a[1]; t[2] = a[2]; t[3] = a[3];
  t[4] = b[0]; t[5] = b[1]; t[6] = b[2]; t[7] = b[3];
  return __builtin_bit_cast(bf16x16, t);
}

// A fragment 16x32 (MxK), element (m,k) at src[m*ld + k].
__device__ __forceinline__ bf16x16 load_a(const unsigned short* src, int ld, int lane) {
  int r = lane & 15, hi = lane >> 4;
  const unsigned short* p = src + r * ld + hi * 8;
  return frag_ld(p, p + 16);
}

// B fragment 32x16 (KxN), element (k,n) at src[n*ld + k] (two 16B runs per lane, k = 8*hi.. and 16+8*hi..).
__device__ __forceinline__ bf16x16 load_b_kp(const unsigned short* src, int ld, int lane) {
  int n = lane & 15, hi = lane >> 4;
  const unsigned short* p = src + n * ld + hi * 8;
  return frag_ld(p, p + 16);
}

// ---------------------------------------------------------------------------
// Kernel 1: fused QKV projection.  x[16384x1024] f32 -> q,k (bf16 row-major),
// vT (bf16 [B][64][T]).  1/sqrt(64)*log2(e) folded into q.
// ---------------------------------------------------------------------------
#define K1_ROWS 128
#define KCH     64
#define XS_LD   72
#define WT_LD   72
#define QSCALE  (0.125f * 1.44269504088896340736f)

__global__ __launch_bounds__(256) void qkv_kernel(
    const float* __restrict__ x, const float* __restrict__ Wq,
    const float* __restrict__ Wk, const float* __restrict__ Wv,
    unsigned short* __restrict__ q_ws, unsigned short* __restrict__ k_ws,
    unsigned short* __restrict__ vT_ws) {
  __shared__ unsigned short xs[K1_ROWS * XS_LD];  // 18 KB
  __shared__ unsigned short wshT[192 * WT_LD];    // 27 KB, [n][k] transposed

  const int tid  = threadIdx.x;
  const int lane = tid & 31, wave = tid >> 5;
  const int hi = lane >> 4, ln = lane & 15;
  const int row0 = blockIdx.x * K1_ROWS;

  f32x8 acc[12];
#pragma unroll
  for (int t = 0; t < 12; ++t) acc[t] = zero8();

  for (int kc = 0; kc < CC; kc += KCH) {
    // stage x tile: float4 reads, packed bf16x4 (b64) LDS stores
#pragma unroll
    for (int j = 0; j < (K1_ROWS * KCH) / (256 * 4); ++j) {  // 8
      int e = (tid + j * 256) * 4;
      int r = e >> 6, c = e & 63;
      float4 xv = *(const float4*)(x + (size_t)(row0 + r) * CC + kc + c);
      u32x2 pk; pk[0] = pack2(xv.x, xv.y); pk[1] = pack2(xv.z, xv.w);
      *(u32x2*)(xs + r * XS_LD + c) = pk;
    }
    // stage W chunk transposed: coalesced float4 reads, u16 transpose stores
#pragma unroll
    for (int j = 0; j < (KCH * 192) / (256 * 4); ++j) {  // 12
      int e = (tid + j * 256) * 4;
      int n = e % 192, k = e / 192;
      const float* w = (n < 64) ? Wq : (n < 128 ? Wk : Wv);
      float4 wv = *(const float4*)(w + (size_t)(kc + k) * HSZ + (n & 63));
      store2bf(&wshT[(n + 0) * WT_LD + k], &wshT[(n + 1) * WT_LD + k], wv.x, wv.y);
      store2bf(&wshT[(n + 2) * WT_LD + k], &wshT[(n + 3) * WT_LD + k], wv.z, wv.w);
    }
    __syncthreads();

    const unsigned short* xrow = xs + wave * 16 * XS_LD;
    bf16x16 a0 = load_a(xrow, XS_LD, lane);       // K = 0..31
    bf16x16 a1 = load_a(xrow + 32, XS_LD, lane);  // K = 32..63

    // software-pipelined B fragments: next tile's loads issue before this WMMA
    bf16x16 b0 = load_b_kp(wshT, WT_LD, lane);
    bf16x16 b1 = load_b_kp(wshT + 32, WT_LD, lane);
#pragma unroll
    for (int t = 0; t < 12; ++t) {
      bf16x16 nb0 = b0, nb1 = b1;
      if (t < 11) {
        const unsigned short* wb = wshT + ((t + 1) * 16) * WT_LD;
        nb0 = load_b_kp(wb, WT_LD, lane);
        nb1 = load_b_kp(wb + 32, WT_LD, lane);
      }
      acc[t] = wmma_bf16(a0, b0, acc[t]);
      acc[t] = wmma_bf16(a1, b1, acc[t]);
      b0 = nb0; b1 = nb1;
    }
    __syncthreads();
  }

  // --- epilogue: all surfaces written wide via per-wave LDS slab -----------
  unsigned short* slab = xs + wave * 1024;  // 2 KB per wave; all waves synced
  const int tile0 = row0 + wave * 16;

  // q (scaled) then k: slab is [16 rows][64 cols], then b128 global stores
#pragma unroll
  for (int pass = 0; pass < 2; ++pass) {
    const float sc = (pass == 0) ? QSCALE : 1.f;
#pragma unroll
    for (int t = 0; t < 4; ++t)
#pragma unroll
      for (int g = 0; g < 8; g += 2)
        store2bf(&slab[(g + 8 * hi) * 64 + t * 16 + ln],
                 &slab[(g + 1 + 8 * hi) * 64 + t * 16 + ln],
                 acc[pass * 4 + t][g] * sc, acc[pass * 4 + t][g + 1] * sc);
    asm volatile("s_wait_dscnt 0x0" ::: "memory");
    unsigned short* dstb = (pass == 0 ? q_ws : k_ws) + (size_t)tile0 * HSZ;
    int r = lane >> 1, half = lane & 1;
#pragma unroll
    for (int j = 0; j < 4; ++j) {
      u32x4 d = *(const u32x4*)(slab + r * 64 + half * 32 + j * 8);
      *(u32x4*)(dstb + (size_t)r * HSZ + half * 32 + j * 8) = d;
    }
    asm volatile("s_wait_dscnt 0x0" ::: "memory");
  }

  // v: slab is [64 h][16 rows] (pairs g,g+1 adjacent -> b32 LDS stores),
  // then contiguous-t b128 global stores into vT
#pragma unroll
  for (int t = 8; t < 12; ++t)
#pragma unroll
    for (int g = 0; g < 8; g += 2) {
      int h = (t - 8) * 16 + ln;
      *(unsigned*)(slab + h * 16 + g + 8 * hi) = pack2(acc[t][g], acc[t][g + 1]);
    }
  asm volatile("s_wait_dscnt 0x0" ::: "memory");
  {
    int bb = tile0 >> 12, trow0 = tile0 & (TT - 1);
#pragma unroll
    for (int j = 0; j < 2; ++j) {
      int h = lane * 2 + j;
      u32x4 d0 = *(const u32x4*)(slab + h * 16);
      u32x4 d1 = *(const u32x4*)(slab + h * 16 + 8);
      unsigned short* dst = vT_ws + ((size_t)bb * HSZ + h) * TT + trow0;
      *(u32x4*)dst = d0;
      *(u32x4*)(dst + 8) = d1;
    }
  }
}

// ---------------------------------------------------------------------------
// Kernel 2: flash attention, one wave per 16-query tile, 64-key blocks.
// ---------------------------------------------------------------------------
__device__ __forceinline__ float redmax16(float x) {
#pragma unroll
  for (int o = 1; o < 16; o <<= 1) x = fmaxf(x, __shfl_xor(x, o, 32));
  return x;
}
__device__ __forceinline__ float redsum16(float x) {
#pragma unroll
  for (int o = 1; o < 16; o <<= 1) x += __shfl_xor(x, o, 32);
  return x;
}

__global__ __launch_bounds__(256) void attn_kernel(
    const unsigned short* __restrict__ q_ws, const unsigned short* __restrict__ k_ws,
    const unsigned short* __restrict__ vT_ws, float* __restrict__ out) {
  __shared__ unsigned short plds[8][16 * 64];  // per-wave private P slab (16 KB)

  const int tid = threadIdx.x, lane = tid & 31, wave = tid >> 5;
  const int hi = lane >> 4, ln = lane & 15;
  const int qt  = blockIdx.x * 8 + wave;
  const int qm0 = qt * 16;
  const int bt  = qm0 >> 12;
  const int t0  = qm0 & (TT - 1);

  const unsigned short* kbase = k_ws + ((size_t)bt << 12) * HSZ;
  const unsigned short* vbase = vT_ws + (size_t)bt * HSZ * TT;

  bf16x16 qa0 = load_a(q_ws + (size_t)qm0 * HSZ, HSZ, lane);
  bf16x16 qa1 = load_a(q_ws + (size_t)qm0 * HSZ + 32, HSZ, lane);

  f32x8 o_acc[4];
  float m[8], l[8];
#pragma unroll
  for (int i = 0; i < 4; ++i) o_acc[i] = zero8();
#pragma unroll
  for (int g = 0; g < 8; ++g) { m[g] = -1e30f; l[g] = 0.f; }

  unsigned short* myp = &plds[wave][0];
  const int nkb = (t0 >> 6) + 1;

  for (int kbi = 0; kbi < nkb; ++kbi) {
    const int kb = kbi << 6;
    __builtin_prefetch(kbase + (size_t)(kb + 64) * HSZ, 0, 1);  // global_prefetch_b8

    // S = q . k^T  — K fragments double-buffered
    f32x8 s[4];
    bf16x16 kf0 = load_b_kp(kbase + (size_t)kb * HSZ, HSZ, lane);
    bf16x16 kf1 = load_b_kp(kbase + (size_t)kb * HSZ + 32, HSZ, lane);
#pragma unroll
    for (int nt = 0; nt < 4; ++nt) {
      bf16x16 nk0 = kf0, nk1 = kf1;
      if (nt < 3) {
        const unsigned short* kp = kbase + (size_t)(kb + (nt + 1) * 16) * HSZ;
        nk0 = load_b_kp(kp, HSZ, lane);
        nk1 = load_b_kp(kp + 32, HSZ, lane);
      }
      s[nt] = wmma_bf16(qa0, kf0, zero8());
      s[nt] = wmma_bf16(qa1, kf1, s[nt]);
      kf0 = nk0; kf1 = nk1;
    }

    // issue first V fragments now — latency hidden under the softmax VALU chain
    bf16x16 vv0 = load_b_kp(vbase + kb, TT, lane);
    bf16x16 vv1 = load_b_kp(vbase + kb + 32, TT, lane);

    // causal mask (only blocks straddling the diagonal)
    if (kb + 63 > t0) {
#pragma unroll
      for (int nt = 0; nt < 4; ++nt) {
        int key = kb + nt * 16 + ln;
#pragma unroll
        for (int g = 0; g < 8; ++g) {
          int row = t0 + g + 8 * hi;
          if (key > row) s[nt][g] = -1e30f;
        }
      }
    }

    // online softmax in C/D layout (row = g + 8*hi, col = lane%16)
#pragma unroll
    for (int g = 0; g < 8; ++g) {
      float rm = fmaxf(fmaxf(s[0][g], s[1][g]), fmaxf(s[2][g], s[3][g]));
      rm = redmax16(rm);
      float mn = fmaxf(m[g], rm);
      float alpha = __builtin_amdgcn_exp2f(m[g] - mn);
      m[g] = mn;
      float rs = 0.f;
#pragma unroll
      for (int nt = 0; nt < 4; ++nt) {
        float p = __builtin_amdgcn_exp2f(s[nt][g] - mn);
        s[nt][g] = p;
        rs += p;
      }
      rs = redsum16(rs);
      l[g] = l[g] * alpha + rs;
#pragma unroll
      for (int nh = 0; nh < 4; ++nh) o_acc[nh][g] *= alpha;
    }

    // P: C-layout f32 -> LDS bf16 (row-major) -> A-layout frags via b128 loads
#pragma unroll
    for (int nt = 0; nt < 4; ++nt)
#pragma unroll
      for (int g = 0; g < 8; g += 2)
        store2bf(&myp[(g + 8 * hi) * 64 + nt * 16 + ln],
                 &myp[(g + 1 + 8 * hi) * 64 + nt * 16 + ln],
                 s[nt][g], s[nt][g + 1]);
    asm volatile("s_wait_dscnt 0x0" ::: "memory");
    bf16x16 pa0 = load_a(myp, 64, lane);
    bf16x16 pa1 = load_a(myp + 32, 64, lane);

    // O += P . V — V fragments double-buffered
#pragma unroll
    for (int nh = 0; nh < 4; ++nh) {
      bf16x16 nv0 = vv0, nv1 = vv1;
      if (nh < 3) {
        const unsigned short* vp = vbase + (size_t)((nh + 1) * 16) * TT + kb;
        nv0 = load_b_kp(vp, TT, lane);
        nv1 = load_b_kp(vp + 32, TT, lane);
      }
      o_acc[nh] = wmma_bf16(pa0, vv0, o_acc[nh]);
      o_acc[nh] = wmma_bf16(pa1, vv1, o_acc[nh]);
      vv0 = nv0; vv1 = nv1;
    }
  }

  // normalize and store fp32
#pragma unroll
  for (int g = 0; g < 8; ++g) {
    float inv = 1.f / l[g];
#pragma unroll
    for (int nh = 0; nh < 4; ++nh)
      out[(size_t)(qm0 + g + 8 * hi) * HSZ + nh * 16 + ln] = o_acc[nh][g] * inv;
  }
}

extern "C" void kernel_launch(void* const* d_in, const int* in_sizes, int n_in,
                              void* d_out, int out_size, void* d_ws, size_t ws_size,
                              hipStream_t stream) {
  const float* x  = (const float*)d_in[0];
  const float* Wq = (const float*)d_in[1];
  const float* Wk = (const float*)d_in[2];
  const float* Wv = (const float*)d_in[3];

  unsigned short* q_ws = (unsigned short*)d_ws;                 // 2 MB
  unsigned short* k_ws = q_ws + (size_t)NROWS * HSZ;            // 2 MB
  unsigned short* vT   = k_ws + (size_t)NROWS * HSZ;            // 2 MB
  float* out = (float*)d_out;

  qkv_kernel<<<NROWS / K1_ROWS, 256, 0, stream>>>(x, Wq, Wk, Wv, q_ws, k_ws, vT);
  attn_kernel<<<(NROWS / 16) / 8, 256, 0, stream>>>(q_ws, k_ws, vT, out);
}
